// GCN_V_UDA_63745904607631
// MI455X (gfx1250) — matrix operand order repaired and checked by reading, not compiled, not run on hardware
//
#include <hip/hip_runtime.h>

typedef __attribute__((ext_vector_type(16))) __bf16 v16bf;
typedef __attribute__((ext_vector_type(8)))  __bf16 v8bf;
typedef __attribute__((ext_vector_type(4)))  __bf16 v4bf;
typedef __attribute__((ext_vector_type(8)))  float  v8f;
typedef __attribute__((ext_vector_type(2)))  float  v2f;

#define D_DIM 2048
#define K_NN  64
#define B_SZ  64
#define H_DIM 1024
#define ROWS  (B_SZ * K_NN)      /* 4096 query rows (b*K + k) */

// ---------------------------------------------------------------------------
// helpers
// ---------------------------------------------------------------------------
__device__ __forceinline__ void splitf(float v, __bf16& h, __bf16& l) {
  h = (__bf16)v;
  l = (__bf16)(v - (float)h);
}
__device__ __forceinline__ v16bf cat8(v8bf a, v8bf b) {
  v16bf r;
#pragma unroll
  for (int i = 0; i < 8; ++i) { r[i] = a[i]; r[i + 8] = b[i]; }
  return r;
}
__device__ __forceinline__ v8f wmma3(v16bf ah, v16bf al, v16bf bh, v16bf bl, v8f acc) {
  acc = __builtin_amdgcn_wmma_f32_16x16x32_bf16(false, ah, false, bh, (short)0, acc, false, false);
  acc = __builtin_amdgcn_wmma_f32_16x16x32_bf16(false, al, false, bh, (short)0, acc, false, false);
  acc = __builtin_amdgcn_wmma_f32_16x16x32_bf16(false, ah, false, bl, (short)0, acc, false, false);
  return acc;
}

// ---------------------------------------------------------------------------
// 0) pack weights: W [Ktot x N] f32 -> bf16 hi/lo in per-lane WMMA B layout.
//    packed[( (tn*KC + kc)*32 + lane )*16 + e] = W[kc*32 + 8*(lane>>4) + (e&7) + 16*(e>>3)][tn*16 + (lane&15)]
// ---------------------------------------------------------------------------
__global__ void pack_weight_kernel(const float* __restrict__ W, int Ktot, int Ncol,
                                   __bf16* __restrict__ Wph, __bf16* __restrict__ Wpl) {
  const int tid  = blockIdx.x * 256 + threadIdx.x;   // one packed element
  const int e    = tid & 15;
  const int lane = (tid >> 4) & 31;
  const int rest = tid >> 9;
  const int KC   = Ktot >> 5;
  const int kc   = rest & (KC - 1);                  // KC is a power of two
  const int tn   = rest / KC;
  const int k    = kc * 32 + 8 * (lane >> 4) + (e & 7) + ((e >> 3) << 4);
  const int c    = tn * 16 + (lane & 15);
  __bf16 h, l;
  splitf(W[(size_t)k * Ncol + c], h, l);
  Wph[tid] = h;
  Wpl[tid] = l;
}

// ---------------------------------------------------------------------------
// 1) gather: stage neighbor row through LDS with gfx1250 async loads, then
//    fan out f32 copy + bf16 hi/lo split copies.
// ---------------------------------------------------------------------------
__global__ void gather_kernel(const float* __restrict__ features,
                              const int* __restrict__ knn,
                              float* __restrict__ feat32,
                              __bf16* __restrict__ fhi, __bf16* __restrict__ flo) {
  __shared__ float4 stage[D_DIM / 4];                // 8 KB
  const int row = blockIdx.x;
  const int idx = knn[row];
  const float4* __restrict__ src = (const float4*)(features + (size_t)idx * D_DIM);
  const unsigned lbase = (unsigned)(uintptr_t)(void*)stage;

  for (int i = threadIdx.x; i < D_DIM / 4; i += 256) {
    const unsigned loff = lbase + i * 16;
    const float4* gp = src + i;
    asm volatile("global_load_async_to_lds_b128 %0, %1, off"
                 :: "v"(loff), "v"(gp) : "memory");
  }
  asm volatile("s_wait_asynccnt 0" ::: "memory");
  __syncthreads();

  float4* __restrict__ dst = (float4*)(feat32 + (size_t)row * D_DIM);
  for (int i = threadIdx.x; i < D_DIM / 4; i += 256) {
    float4 v = stage[i];
    dst[i] = v;
    const float vv[4] = {v.x, v.y, v.z, v.w};
    v4bf hv, lv;
#pragma unroll
    for (int j = 0; j < 4; ++j) { __bf16 h, l; splitf(vv[j], h, l); hv[j] = h; lv[j] = l; }
    *(v4bf*)(fhi + (size_t)row * D_DIM + i * 4) = hv;
    *(v4bf*)(flo + (size_t)row * D_DIM + i * 4) = lv;
  }
}

// ---------------------------------------------------------------------------
// 2) Gram + row softmax per batch (f32 WMMA, K=2048). One block / batch.
// ---------------------------------------------------------------------------
__global__ void gram_softmax_kernel(const float* __restrict__ feat,
                                    float* __restrict__ adj) {
  __shared__ float s[64][64];
  const int b    = blockIdx.x;
  const int lane = threadIdx.x & 31;
  const int wave = threadIdx.x >> 5;
  const int hi   = lane >> 4;
  const int n    = lane & 15;
  const int tm   = wave >> 2, tn = wave & 3;

  const float* __restrict__ fb   = feat + (size_t)b * K_NN * D_DIM;
  const float* __restrict__ arow = fb + (size_t)(16 * tm + n) * D_DIM;
  const float* __restrict__ brow = fb + (size_t)(16 * tn + n) * D_DIM;

  v8f acc = {};
  for (int k = 0; k < D_DIM; k += 4) {
    v2f a, bm;
    a[0]  = arow[k + 2 * hi];
    a[1]  = arow[k + 2 * hi + 1];
    bm[0] = brow[k + 2 * hi];
    bm[1] = brow[k + 2 * hi + 1];
    acc = __builtin_amdgcn_wmma_f32_16x16x4_f32(false, a, false, bm,
                                                (short)0, acc, false, false);
  }
#pragma unroll
  for (int r = 0; r < 8; ++r)
    s[16 * tm + r + 8 * hi][16 * tn + n] = acc[r];
  __syncthreads();

  if (threadIdx.x < 64) {
    const int row = threadIdx.x;
    float m = s[row][0];
#pragma unroll 4
    for (int j = 1; j < 64; ++j) m = fmaxf(m, s[row][j]);
    float sum = 0.f;
#pragma unroll 4
    for (int j = 0; j < 64; ++j) { float e = __expf(s[row][j] - m); s[row][j] = e; sum += e; }
    const float inv = 1.f / sum;
    float* __restrict__ ab = adj + (size_t)b * 4096 + (size_t)row * 64;
#pragma unroll 4
    for (int j = 0; j < 64; ++j) ab[j] = s[row][j] * inv;
  }
}

// ---------------------------------------------------------------------------
// 3) agg = adj @ feat (f32 WMMA), epilogue emits bf16 hi/lo directly.
// ---------------------------------------------------------------------------
__global__ void agg_kernel(const float* __restrict__ adj,
                           const float* __restrict__ feat,
                           __bf16* __restrict__ ghi, __bf16* __restrict__ glo) {
  const int b    = blockIdx.y;
  const int lane = threadIdx.x & 31;
  const int wave = threadIdx.x >> 5;
  const int hi   = lane >> 4;
  const int n    = lane & 15;
  const int t    = blockIdx.x * 16 + wave;   // tile 0..511
  const int tm   = t >> 7, tn = t & 127;

  const float* __restrict__ ab   = adj + (size_t)b * 4096;
  const float* __restrict__ fb   = feat + (size_t)b * K_NN * D_DIM;
  const float* __restrict__ arow = ab + (size_t)(16 * tm + n) * 64;

  v8f acc = {};
  for (int k = 0; k < 64; k += 4) {
    v2f a, bm;
    a[0]  = arow[k + 2 * hi];
    a[1]  = arow[k + 2 * hi + 1];
    bm[0] = fb[(size_t)(k + 2 * hi)     * D_DIM + 16 * tn + n];
    bm[1] = fb[(size_t)(k + 2 * hi + 1) * D_DIM + 16 * tn + n];
    acc = __builtin_amdgcn_wmma_f32_16x16x4_f32(false, a, false, bm,
                                                (short)0, acc, false, false);
  }
  const size_t obase = (size_t)b * K_NN * D_DIM;
#pragma unroll
  for (int r = 0; r < 8; ++r) {
    const size_t o = obase + (size_t)(16 * tm + r + 8 * hi) * D_DIM + 16 * tn + n;
    __bf16 h, l;
    splitf(acc[r], h, l);
    ghi[o] = h;
    glo[o] = l;
  }
}

// ---------------------------------------------------------------------------
// bf16 two-term-split GEMM, pre-split/pre-packed operands.
// Block 256 thr = 8 waves (2 m-tiles x 4 n-groups); each wave: 16x64.
// MODE 0: relu(acc+bias) -> bf16 hi/lo outputs   MODE 1: prelu -> f32 output
// ---------------------------------------------------------------------------
template <int MODE>
__global__ void gemm_bf16packed_kernel(
    const __bf16* __restrict__ Ah0, const __bf16* __restrict__ Al0,
    const __bf16* __restrict__ Ah1, const __bf16* __restrict__ Al1,  // optional phase 2
    int Kper,
    const __bf16* __restrict__ Wph, const __bf16* __restrict__ Wpl, int KCtot,
    const float* __restrict__ bias, const float* __restrict__ alpha,
    float* __restrict__ outF, __bf16* __restrict__ outH, __bf16* __restrict__ outL,
    int ldo) {
  const int lane = threadIdx.x & 31;
  const int wave = threadIdx.x >> 5;
  const int hi   = lane >> 4;
  const int n    = lane & 15;
  const int wm   = wave >> 2, wn = wave & 3;
  const int row0 = blockIdx.y * 32 + wm * 16;
  const int tnG0 = blockIdx.x * 16 + wn * 4;   // global 16-col tile index

  v8f acc[4] = {};

  const int phases = Ah1 ? 2 : 1;
  for (int ph = 0; ph < phases; ++ph) {
    const __bf16* __restrict__ Ah = (ph ? Ah1 : Ah0) + (size_t)(row0 + n) * Kper;
    const __bf16* __restrict__ Al = (ph ? Al1 : Al0) + (size_t)(row0 + n) * Kper;
    const int kcBase = (ph * Kper) >> 5;
    for (int kb = 0; kb < Kper; kb += 32) {
      const int kc = kcBase + (kb >> 5);
      __builtin_prefetch(Ah + kb + 128, 0, 0);   // speculative stream-ahead
      // A 16x32 bf16 layout: half-wave 0 needs k {0-7,16-23}, half-wave 1 {8-15,24-31}
      const v16bf ah = cat8(*(const v8bf*)(Ah + kb + 8 * hi),
                            *(const v8bf*)(Ah + kb + 16 + 8 * hi));
      const v16bf al = cat8(*(const v8bf*)(Al + kb + 8 * hi),
                            *(const v8bf*)(Al + kb + 16 + 8 * hi));
#pragma unroll
      for (int t = 0; t < 4; ++t) {
        const size_t bbase = ((size_t)(tnG0 + t) * KCtot + kc) * 512 + lane * 16;
        const v16bf bh = *(const v16bf*)(Wph + bbase);
        const v16bf bl = *(const v16bf*)(Wpl + bbase);
        acc[t] = wmma3(ah, al, bh, bl, acc[t]);
      }
    }
  }

#pragma unroll
  for (int t = 0; t < 4; ++t) {
    const int col = (tnG0 + t) * 16 + n;
    const float bv = bias[col];
    const float av = (MODE == 1) ? alpha[col] : 0.f;
#pragma unroll
    for (int r = 0; r < 8; ++r) {
      const int row = row0 + r + 8 * hi;
      float v = acc[t][r] + bv;
      if (MODE == 0) {
        v = (v > 0.f) ? v : 0.f;
        __bf16 h, l;
        splitf(v, h, l);
        outH[(size_t)row * ldo + col] = h;
        outL[(size_t)row * ldo + col] = l;
      } else {
        v = (v > 0.f) ? v : av * v;
        outF[(size_t)row * ldo + col] = v;
      }
    }
  }
}

// ---------------------------------------------------------------------------
// 6) pred[r] = h[r,:] . Wc2 + bc2    (one block per row)
// ---------------------------------------------------------------------------
__global__ void matvec_kernel(const float* __restrict__ Hm,
                              const float* __restrict__ w,
                              const float* __restrict__ b,
                              float* __restrict__ out) {
  __shared__ float red[256];
  const float* __restrict__ hr = Hm + (size_t)blockIdx.x * H_DIM;
  float s = 0.f;
  for (int i = threadIdx.x; i < H_DIM; i += 256) s += hr[i] * w[i];
  red[threadIdx.x] = s;
  __syncthreads();
  for (int off = 128; off > 0; off >>= 1) {
    if (threadIdx.x < off) red[threadIdx.x] += red[threadIdx.x + off];
    __syncthreads();
  }
  if (threadIdx.x == 0) out[blockIdx.x] = red[0] + b[0];
}

// ---------------------------------------------------------------------------
extern "C" void kernel_launch(void* const* d_in, const int* in_sizes, int n_in,
                              void* d_out, int out_size, void* d_ws, size_t ws_size,
                              hipStream_t stream) {
  const float* features = (const float*)d_in[0];   // [N, D]
  const int*   knn      = (const int*)  d_in[1];   // [B, K]
  const float* W1       = (const float*)d_in[2];   // [2D, H]
  const float* b1       = (const float*)d_in[3];   // [H]
  const float* Wc1      = (const float*)d_in[4];   // [H, H]
  const float* bc1      = (const float*)d_in[5];   // [H]
  const float* alpha    = (const float*)d_in[6];   // [H]
  const float* Wc2      = (const float*)d_in[7];   // [H, 1]
  const float* bc2      = (const float*)d_in[8];   // [1]
  float* out = (float*)d_out;                      // [B, K, 1] = 4096

  char* p = (char*)d_ws;
  float*  feat32 = (float*) p;  p += (size_t)ROWS * D_DIM * 4;   // 33.55 MB
  __bf16* fhi    = (__bf16*)p;  p += (size_t)ROWS * D_DIM * 2;
  __bf16* flo    = (__bf16*)p;  p += (size_t)ROWS * D_DIM * 2;
  __bf16* ghi    = (__bf16*)p;  p += (size_t)ROWS * D_DIM * 2;
  __bf16* glo    = (__bf16*)p;  p += (size_t)ROWS * D_DIM * 2;
  __bf16* w1h    = (__bf16*)p;  p += (size_t)2 * D_DIM * H_DIM * 2;
  __bf16* w1l    = (__bf16*)p;  p += (size_t)2 * D_DIM * H_DIM * 2;
  __bf16* wch    = (__bf16*)p;  p += (size_t)H_DIM * H_DIM * 2;
  __bf16* wcl    = (__bf16*)p;  p += (size_t)H_DIM * H_DIM * 2;
  float*  adj    = (float*) p;  p += (size_t)B_SZ * K_NN * K_NN * 4;
  __bf16* xh     = (__bf16*)p;  p += (size_t)ROWS * H_DIM * 2;
  __bf16* xl     = (__bf16*)p;  p += (size_t)ROWS * H_DIM * 2;
  float*  h      = feat32;      // overlay: feat32 dead after agg kernel

  // 0) pre-split weights into packed bf16 hi/lo (B-matrix lane layout)
  pack_weight_kernel<<<(2 * D_DIM * H_DIM) / 256, 256, 0, stream>>>(W1, 2 * D_DIM, H_DIM, w1h, w1l);
  pack_weight_kernel<<<(H_DIM * H_DIM) / 256, 256, 0, stream>>>(Wc1, H_DIM, H_DIM, wch, wcl);

  // 1) gather neighbor features (async-to-LDS staging) + bf16 split
  gather_kernel<<<ROWS, 256, 0, stream>>>(features, knn, feat32, fhi, flo);

  // 2) Gram + softmax  (f32 WMMA)
  gram_softmax_kernel<<<B_SZ, 512, 0, stream>>>(feat32, adj);

  // 3) agg = adj @ feat  (f32 WMMA) -> bf16 hi/lo
  agg_kernel<<<dim3(32, B_SZ), 512, 0, stream>>>(adj, feat32, ghi, glo);

  // 4) x = relu([feat|agg] @ W1 + b1)   (bf16-split WMMA, K = 2*2048) -> bf16 hi/lo
  gemm_bf16packed_kernel<0><<<dim3(H_DIM / 256, ROWS / 32), 256, 0, stream>>>(
      fhi, flo, ghi, glo, D_DIM, w1h, w1l, (2 * D_DIM) / 32,
      b1, nullptr, nullptr, xh, xl, H_DIM);

  // 5) h = prelu(x @ Wc1 + bc1)         (bf16-split WMMA, K = 1024) -> f32
  gemm_bf16packed_kernel<1><<<dim3(H_DIM / 256, ROWS / 32), 256, 0, stream>>>(
      xh, xl, nullptr, nullptr, H_DIM, wch, wcl, H_DIM / 32,
      bc1, alpha, h, nullptr, nullptr, H_DIM);

  // 6) pred = h @ Wc2 + bc2
  matvec_kernel<<<ROWS, 256, 0, stream>>>(h, Wc2, bc2, out);
}
